// Model_58222576665013
// MI455X (gfx1250) — compile-verified, matrix operand order checked by reading, not run on hardware
//
#include <hip/hip_runtime.h>
#include <hip/hip_bf16.h>
#include <math.h>
#include <stdint.h>

#define Bq 16
#define Lq 512
#define Cq 321
#define PLq 96
#define Pq 16
#define Sq 8
#define Kq 8
#define Dq 256
#define PNq 64
#define NSq (Bq*Cq)      /* 5136 samples */
#define KDq (PNq*Dq)     /* 16384 */
#define LPq (Lq+Sq)      /* 520 padded length */

typedef _Float16 h8  __attribute__((ext_vector_type(8)));
typedef _Float16 h16 __attribute__((ext_vector_type(16)));
typedef float    f8  __attribute__((ext_vector_type(8)));
typedef unsigned int u32x4 __attribute__((ext_vector_type(4)));
typedef int          i32x4 __attribute__((ext_vector_type(4)));
typedef int          i32x8 __attribute__((ext_vector_type(8)));

union Frag { h16 v; h8 h[2]; };

__device__ __forceinline__ float gelu_f(float x) {
  return 0.5f * x * (1.0f + erff(x * 0.70710678118654752440f));
}

__device__ __forceinline__ f8 wmma16(h16 a, h16 b, f8 c) {
  // D = A(16x32 f16) * B(32x16 f16) + C(16x16 f32)
  return __builtin_amdgcn_wmma_f32_16x16x32_f16(false, a, false, b, (short)0, c,
                                                false, false);
}

// ---------------------------------------------------------------------------
// Tensor Data Mover: DMA a 2D f16 tile (64 x nrows, row stride = krow elems)
// from global into LDS (row-major, 64 halves per row). D# per ISA 08 §8.3/8.4.
// ---------------------------------------------------------------------------
__device__ __forceinline__ void tdm_load_tile64(uint32_t lds_addr,
                                                const _Float16* gptr,
                                                uint32_t krow, uint32_t nrows) {
  uint64_t ga = (uint64_t)(uintptr_t)gptr;
  u32x4 g0;
  g0[0] = 1u;                                   // count=1, user descriptor
  g0[1] = lds_addr;                             // lds_addr [63:32]
  g0[2] = (uint32_t)ga;                         // global_addr [95:64]
  g0[3] = ((uint32_t)(ga >> 32) & 0x01FFFFFFu)  // global_addr [120:96]
          | 0x80000000u;                        // type=2 ("image") [127:126]
  i32x8 g1;
  g1[0] = (int)(1u << 16);                      // data_size=1 -> 2 bytes
  g1[1] = (int)((krow & 0xFFFFu) << 16);        // tensor_dim0[15:0] @ [63:48]
  g1[2] = (int)(((krow >> 16) & 0xFFFFu)        // tensor_dim0[31:16]
          | ((nrows & 0xFFFFu) << 16));         // tensor_dim1[15:0]
  g1[3] = (int)(((nrows >> 16) & 0xFFFFu)       // tensor_dim1[31:16]
          | (64u << 16));                       // tile_dim0 = 64 elems
  g1[4] = (int)(nrows & 0xFFFFu);               // tile_dim1 = nrows, tile_dim2=0
  g1[5] = (int)krow;                            // tensor_dim0_stride[31:0]
  g1[6] = 0;                                    // stride0[47:32], stride1[15:0]
  g1[7] = 0;
  i32x4 gz; gz[0] = 0; gz[1] = 0; gz[2] = 0; gz[3] = 0;
#if __clang_major__ >= 23
  i32x8 gz8;
  gz8[0]=0; gz8[1]=0; gz8[2]=0; gz8[3]=0; gz8[4]=0; gz8[5]=0; gz8[6]=0; gz8[7]=0;
  __builtin_amdgcn_tensor_load_to_lds(g0, g1, gz, gz, gz8, 0);
#else
  __builtin_amdgcn_tensor_load_to_lds(g0, g1, gz, gz, 0);
#endif
}

// ---------------------------------------------------------------------------
// 1) RevIN statistics: mean / stdev / rstd per (b,c)
// ---------------------------------------------------------------------------
__global__ __launch_bounds__(128) void stats_kernel(
    const float* __restrict__ x, float* __restrict__ meanB,
    float* __restrict__ sdevB, float* __restrict__ rstdB) {
  int b = blockIdx.x;
  int c = blockIdx.y * 128 + threadIdx.x;
  if (c >= Cq) return;
  float s = 0.f, s2 = 0.f;
  for (int l = 0; l < Lq; ++l) {
    float v = x[((size_t)b * Lq + l) * Cq + c];
    s += v; s2 += v * v;
  }
  float m = s * (1.0f / Lq);
  float var = s2 * (1.0f / Lq) - m * m;
  float sd = sqrtf(var + 1e-5f);
  meanB[b * Cq + c] = m;
  sdevB[b * Cq + c] = sd;
  rstdB[b * Cq + c] = 1.0f / sd;
}

// ---------------------------------------------------------------------------
// 2) Normalize -> f16 xh (padded to 520, tail repeats last), fused gate proj g
// ---------------------------------------------------------------------------
__global__ __launch_bounds__(128) void norm_kernel(
    const float* __restrict__ x, const float* __restrict__ meanB,
    const float* __restrict__ rstdB, const float* __restrict__ rw,
    const float* __restrict__ rb, const float* __restrict__ sw,
    const float* __restrict__ sb, _Float16* __restrict__ xh,
    float* __restrict__ g) {
  __shared__ float red[128];
  int bl = blockIdx.x;
  int b = bl >> 9;          // / 512
  int l = bl & (Lq - 1);
  float part = 0.f;
  for (int c = threadIdx.x; c < Cq; c += 128) {
    int n = b * Cq + c;
    float xn = (x[((size_t)b * Lq + l) * Cq + c] - meanB[n]) * rstdB[n] * rw[c] + rb[c];
    xh[(size_t)n * LPq + l] = (_Float16)xn;
    if (l == Lq - 1) {
      for (int t = Lq; t < LPq; ++t) xh[(size_t)n * LPq + t] = (_Float16)xn;
    }
    part += xn * sw[c];
  }
  red[threadIdx.x] = part;
  __syncthreads();
  for (int s = 64; s > 0; s >>= 1) {
    if (threadIdx.x < s) red[threadIdx.x] += red[threadIdx.x + s];
    __syncthreads();
  }
  if (threadIdx.x == 0) g[b * Lq + l] = red[0] + sb[0];
}

// ---------------------------------------------------------------------------
// 3) Gating: logits, top-k, softmax, gates-sum, balance scalar
// ---------------------------------------------------------------------------
__device__ float cv2_4(const float* xv) {
  float m = (xv[0] + xv[1] + xv[2] + xv[3]) * 0.25f;
  float var = 0.f;
  for (int i = 0; i < 4; ++i) { float d = xv[i] - m; var += d * d; }
  var *= (1.0f / 3.0f);  // ddof=1
  return var / (m * m + 1e-10f);
}

__global__ void gate_kernel(const float* __restrict__ g,
                            const float* __restrict__ gate_w,
                            const float* __restrict__ gate_b,
                            float* __restrict__ gsum,
                            float* __restrict__ balance_out) {
  __shared__ float gates[Bq][4];
  int b = threadIdx.x;
  if (b < Bq) {
    float lg[4];
    for (int e = 0; e < 4; ++e) {
      float s = gate_b[e];
      for (int l = 0; l < Lq; ++l) s += g[b * Lq + l] * gate_w[e * Lq + l];
      lg[e] = s;
    }
    int idx[4] = {0, 1, 2, 3};
    for (int i = 0; i < 3; ++i) {
      int mi = i;
      for (int j = i + 1; j < 4; ++j)
        if (lg[idx[j]] > lg[idx[mi]]) mi = j;
      int t = idx[i]; idx[i] = idx[mi]; idx[mi] = t;
    }
    float v0 = lg[idx[0]], v1 = lg[idx[1]];
    float mx = fmaxf(v0, v1);
    float e0 = expf(v0 - mx), e1 = expf(v1 - mx);
    float den = e0 + e1;
    for (int e = 0; e < 4; ++e) gates[b][e] = 0.f;
    gates[b][idx[0]] = e0 / den;
    gates[b][idx[1]] = e1 / den;
    gsum[b] = (e0 + e1) / den;
  }
  __syncthreads();
  if (threadIdx.x == 0) {
    float imp[4], ld[4];
    for (int e = 0; e < 4; ++e) {
      float s = 0.f, c = 0.f;
      for (int bb = 0; bb < Bq; ++bb) {
        s += gates[bb][e];
        c += (gates[bb][e] > 0.f) ? 1.f : 0.f;
      }
      imp[e] = s; ld[e] = c;
    }
    balance_out[0] = (cv2_4(imp) + cv2_4(ld)) * 0.01f;
  }
}

// ---------------------------------------------------------------------------
// 4) f32 -> f16 convert
// ---------------------------------------------------------------------------
__global__ __launch_bounds__(256) void cvt_kernel(const float* __restrict__ src,
                                                  _Float16* __restrict__ dst,
                                                  int n) {
  int i = blockIdx.x * 256 + threadIdx.x;
  if (i < n) dst[i] = (_Float16)src[i];
}

// ---------------------------------------------------------------------------
// 5) Patch embedding z = patches(64x16) @ wp_w^T(16x256) + wp_b  (WMMA, K pad)
// ---------------------------------------------------------------------------
__global__ __launch_bounds__(128) void z_kernel(
    const _Float16* __restrict__ xh, const _Float16* __restrict__ wp16,
    const float* __restrict__ wp_b, _Float16* __restrict__ z) {
  __shared__ __align__(16) _Float16 rowbuf[LPq + 8];
  int n = blockIdx.x;
  for (int i = threadIdx.x; i < LPq; i += 128) rowbuf[i] = xh[(size_t)n * LPq + i];
  __syncthreads();

  int wave = threadIdx.x >> 5, lane = threadIdx.x & 31;
  int koff = (lane < 16) ? 0 : 8;
  int r0 = (lane < 16) ? 0 : 8;
  h8 zero8;
#pragma unroll
  for (int i = 0; i < 8; ++i) zero8[i] = (_Float16)0.0f;

  int pn = wave * 16 + (lane & 15);
  Frag a;
  a.h[0] = *(const h8*)(&rowbuf[pn * 8 + koff]);  // K = p (0..7 / 8..15)
  a.h[1] = zero8;                                 // K 16..31 zero-padded

#pragma unroll 4
  for (int nt = 0; nt < 16; ++nt) {
    int d = nt * 16 + (lane & 15);
    Frag bfr;
    bfr.h[0] = *(const h8*)(wp16 + d * Pq + koff);
    bfr.h[1] = zero8;
    f8 acc = {0.f, 0.f, 0.f, 0.f, 0.f, 0.f, 0.f, 0.f};
    acc = wmma16(a.v, bfr.v, acc);
    float bias = wp_b[d];
#pragma unroll
    for (int r = 0; r < 8; ++r) {
      int opn = wave * 16 + r + r0;
      z[(size_t)n * KDq + opn * Dq + d] = (_Float16)(acc[r] + bias);
    }
  }
}

// ---------------------------------------------------------------------------
// 6) Tall-skinny WMMA GEMM: out(MxN) = A(MxK f16) @ Bw(NxK f16)^T
//    B tiles DMAed into double-buffered LDS by the Tensor Data Mover
//    (wave 0 issues, s_wait_tensorcnt + barrier), A streamed from global.
//    MODE 0: +bias -> f32   MODE 1: +bias, GELU -> f16   MODE 2: +bias, += f32
// ---------------------------------------------------------------------------
template <int NTILES, int MODE>
__global__ __launch_bounds__(256) void gemm_nt_kernel(
    const _Float16* __restrict__ A, const _Float16* __restrict__ Bw,
    const float* __restrict__ bias, float* __restrict__ outF,
    _Float16* __restrict__ outH, int M, int K) {
  const int N = NTILES * 16;
  __shared__ __align__(16) _Float16 Bs[2][NTILES * 16 * 64];
  int wave = threadIdx.x >> 5, lane = threadIdx.x & 31;
  int mt = blockIdx.x * 8 + wave;
  bool active = (mt * 16 < M);
  int mtc = active ? mt : (M / 16 - 1);  // clamp: all waves stay for barriers
  int row = mtc * 16 + (lane & 15);
  int koff = (lane < 16) ? 0 : 8;
  int r0 = (lane < 16) ? 0 : 8;

  f8 acc[NTILES];
#pragma unroll
  for (int i = 0; i < NTILES; ++i)
    acc[i] = (f8){0.f, 0.f, 0.f, 0.f, 0.f, 0.f, 0.f, 0.f};

  const _Float16* arow = A + (size_t)row * K + koff;
  uint32_t lds0 = (uint32_t)(uintptr_t)(&Bs[0][0]);
  uint32_t lds1 = (uint32_t)(uintptr_t)(&Bs[1][0]);

  if (wave == 0) tdm_load_tile64(lds0, Bw, (uint32_t)K, (uint32_t)N);

  int cur = 0;
  for (int kk = 0; kk < K; kk += 64) {
    if (wave == 0) {
      if (kk + 64 < K) {
        tdm_load_tile64(cur ? lds0 : lds1, Bw + kk + 64, (uint32_t)K, (uint32_t)N);
        __builtin_amdgcn_s_wait_tensorcnt((short)1);  // prior tile done
      } else {
        __builtin_amdgcn_s_wait_tensorcnt((short)0);
      }
    }
    __syncthreads();  // B tile visible to all waves
    const _Float16* bs = &Bs[cur][0];
    __builtin_prefetch(arow + kk + 2048, 0, 0);  // prefetch A stream
#pragma unroll
    for (int s = 0; s < 64; s += 32) {
      Frag a;
      a.h[0] = *(const h8*)(arow + kk + s);
      a.h[1] = *(const h8*)(arow + kk + s + 16);
#pragma unroll
      for (int nt = 0; nt < NTILES; ++nt) {
        int col = nt * 16 + (lane & 15);
        Frag b;
        b.h[0] = *(const h8*)(bs + col * 64 + s + koff);
        b.h[1] = *(const h8*)(bs + col * 64 + s + koff + 16);
        acc[nt] = wmma16(a.v, b.v, acc[nt]);
      }
    }
    __syncthreads();  // safe to overwrite this buffer next round
    cur ^= 1;
  }

  if (active) {
#pragma unroll
    for (int nt = 0; nt < NTILES; ++nt) {
      int col = nt * 16 + (lane & 15);
      float bv = bias[col];
#pragma unroll
      for (int r = 0; r < 8; ++r) {
        int orow = mt * 16 + r + r0;
        float v = acc[nt][r] + bv;
        if (MODE == 1) {
          outH[(size_t)orow * N + col] = (_Float16)gelu_f(v);
        } else if (MODE == 2) {
          outF[(size_t)orow * N + col] += v;
        } else {
          outF[(size_t)orow * N + col] = v;
        }
      }
    }
  }
}

// ---------------------------------------------------------------------------
// 7) Depthwise conv (K=8, pad 3/4) + GELU + BN1 + residual; one wave per row
// ---------------------------------------------------------------------------
__global__ __launch_bounds__(256) void dwconv_kernel(
    const _Float16* __restrict__ hin, _Float16* __restrict__ hout,
    const float* __restrict__ dw_w, const float* __restrict__ dw_b,
    const float* __restrict__ bng, const float* __restrict__ bnb,
    const float* __restrict__ bnm, const float* __restrict__ bnv) {
  __shared__ __align__(16) _Float16 buf[8][Dq];
  int wave = threadIdx.x >> 5, lane = threadIdx.x & 31;
  int rid = blockIdx.x * 8 + wave;
  int n = rid >> 6, c = rid & 63;
  const _Float16* src = hin + (size_t)n * KDq + c * Dq;
  *(h8*)(&buf[wave][lane * 8]) = *(const h8*)(src + lane * 8);
  __syncthreads();

  float scale = bng[c] * rsqrtf(bnv[c] + 1e-5f);
  float shift = bnb[c] - bnm[c] * scale;
  float w[8];
#pragma unroll
  for (int k = 0; k < 8; ++k) w[k] = dw_w[c * Kq + k];
  float bias = dw_b[c];
  _Float16* dst = hout + (size_t)n * KDq + c * Dq;

#pragma unroll
  for (int j = 0; j < 8; ++j) {
    int t = lane * 8 + j;
    float s = bias;
#pragma unroll
    for (int k = 0; k < 8; ++k) {
      int ti = t + k - 3;
      float xv = (ti >= 0 && ti < Dq) ? (float)buf[wave][ti] : 0.f;
      s += w[k] * xv;
    }
    float o = (float)buf[wave][t] + gelu_f(s) * scale + shift;
    dst[t] = (_Float16)o;
  }
}

// ---------------------------------------------------------------------------
// 8) Pointwise mix: out(64x256) = pw_w(64x64) @ h(64x256), +bias GELU BN2
// ---------------------------------------------------------------------------
__global__ __launch_bounds__(128) void pw_kernel(
    const _Float16* __restrict__ hin, _Float16* __restrict__ hout,
    const _Float16* __restrict__ pwW16, const float* __restrict__ pw_b,
    const float* __restrict__ bng, const float* __restrict__ bnb,
    const float* __restrict__ bnm, const float* __restrict__ bnv) {
  __shared__ __align__(16) _Float16 XT[Dq * PNq];  // XT[t][c], 32KB
  int n = blockIdx.x;
  const _Float16* src = hin + (size_t)n * KDq;
  int tid = threadIdx.x;
  int c = tid & 63;
  int th = tid >> 6;  // 0..1
  for (int t0 = th * 128; t0 < th * 128 + 128; t0 += 8) {
    h8 d = *(const h8*)(src + c * Dq + t0);
#pragma unroll
    for (int j = 0; j < 8; ++j) XT[(t0 + j) * PNq + c] = d[j];
  }
  __syncthreads();

  int wave = tid >> 5, lane = tid & 31;
  int koff = (lane < 16) ? 0 : 8;
  int r0 = (lane < 16) ? 0 : 8;
  int orow = wave * 16 + (lane & 15);
  Frag a0, a1;
  a0.h[0] = *(const h8*)(pwW16 + orow * PNq + koff);
  a0.h[1] = *(const h8*)(pwW16 + orow * PNq + koff + 16);
  a1.h[0] = *(const h8*)(pwW16 + orow * PNq + 32 + koff);
  a1.h[1] = *(const h8*)(pwW16 + orow * PNq + 32 + koff + 16);

  float bias8[8], sc8[8], sh8[8];
#pragma unroll
  for (int r = 0; r < 8; ++r) {
    int o = wave * 16 + r + r0;
    bias8[r] = pw_b[o];
    float s = bng[o] * rsqrtf(bnv[o] + 1e-5f);
    sc8[r] = s;
    sh8[r] = bnb[o] - bnm[o] * s;
  }
  _Float16* dst = hout + (size_t)n * KDq;

#pragma unroll 4
  for (int nt = 0; nt < 16; ++nt) {
    int t = nt * 16 + (lane & 15);
    Frag b0, b1;
    b0.h[0] = *(const h8*)(&XT[t * PNq + koff]);
    b0.h[1] = *(const h8*)(&XT[t * PNq + koff + 16]);
    b1.h[0] = *(const h8*)(&XT[t * PNq + 32 + koff]);
    b1.h[1] = *(const h8*)(&XT[t * PNq + 32 + koff + 16]);
    f8 acc = {0.f, 0.f, 0.f, 0.f, 0.f, 0.f, 0.f, 0.f};
    acc = wmma16(a0.v, b0.v, acc);
    acc = wmma16(a1.v, b1.v, acc);
#pragma unroll
    for (int r = 0; r < 8; ++r) {
      int o = wave * 16 + r + r0;
      float val = gelu_f(acc[r] + bias8[r]) * sc8[r] + sh8[r];
      dst[o * Dq + t] = (_Float16)val;
    }
  }
}

// ---------------------------------------------------------------------------
// 9) Final: y[b,pl,c] = ((uv*gsum - rb)/(rw+1e-10))*stdev + mean
// ---------------------------------------------------------------------------
__global__ __launch_bounds__(256) void final_kernel(
    const float* __restrict__ uv, const float* __restrict__ gsum,
    const float* __restrict__ rw, const float* __restrict__ rb,
    const float* __restrict__ meanB, const float* __restrict__ sdevB,
    float* __restrict__ out) {
  int i = blockIdx.x * 256 + threadIdx.x;
  if (i >= Bq * PLq * Cq) return;
  int c = i % Cq;
  int pl = (i / Cq) % PLq;
  int b = i / (Cq * PLq);
  int n = b * Cq + c;
  float val = uv[(size_t)n * PLq + pl] * gsum[b];
  val = (val - rb[c]) / (rw[c] + 1e-10f);
  out[i] = val * sdevB[n] + meanB[n];
}

// ---------------------------------------------------------------------------
extern "C" void kernel_launch(void* const* d_in, const int* in_sizes, int n_in,
                              void* d_out, int out_size, void* d_ws,
                              size_t ws_size, hipStream_t stream) {
  (void)in_sizes; (void)n_in; (void)out_size; (void)ws_size;
  const float* x       = (const float*)d_in[0];
  const float* revin_w = (const float*)d_in[1];
  const float* revin_b = (const float*)d_in[2];
  const float* start_w = (const float*)d_in[3];
  const float* start_b = (const float*)d_in[4];
  const float* gate_w  = (const float*)d_in[5];
  const float* gate_b  = (const float*)d_in[6];
  const float* wp_w    = (const float*)d_in[7];
  const float* wp_b    = (const float*)d_in[8];
  const float* h0_w    = (const float*)d_in[9];
  const float* h0_b    = (const float*)d_in[10];
  const float* h1_w1   = (const float*)d_in[11];
  const float* h1_b1   = (const float*)d_in[12];
  const float* h1_w2   = (const float*)d_in[13];
  const float* h1_b2   = (const float*)d_in[14];
  const float* dw_w[2]  = {(const float*)d_in[15], (const float*)d_in[27]};
  const float* dw_b[2]  = {(const float*)d_in[16], (const float*)d_in[28]};
  const float* bn1g[2]  = {(const float*)d_in[17], (const float*)d_in[29]};
  const float* bn1b[2]  = {(const float*)d_in[18], (const float*)d_in[30]};
  const float* bn1m[2]  = {(const float*)d_in[19], (const float*)d_in[31]};
  const float* bn1v[2]  = {(const float*)d_in[20], (const float*)d_in[32]};
  const float* pw_w[2]  = {(const float*)d_in[21], (const float*)d_in[33]};
  const float* pw_b[2]  = {(const float*)d_in[22], (const float*)d_in[34]};
  const float* bn2g[2]  = {(const float*)d_in[23], (const float*)d_in[35]};
  const float* bn2b[2]  = {(const float*)d_in[24], (const float*)d_in[36]};
  const float* bn2m[2]  = {(const float*)d_in[25], (const float*)d_in[37]};
  const float* bn2v[2]  = {(const float*)d_in[26], (const float*)d_in[38]};

  char* ws = (char*)d_ws;
  size_t off = 0;
  auto carve = [&](size_t bytes) -> void* {
    void* p = ws + off;
    off = (off + bytes + 255) & ~(size_t)255;
    return p;
  };
  float*     meanB  = (float*)carve((size_t)NSq * 4);
  float*     sdevB  = (float*)carve((size_t)NSq * 4);
  float*     rstdB  = (float*)carve((size_t)NSq * 4);
  _Float16*  xh     = (_Float16*)carve((size_t)NSq * LPq * 2);
  float*     gbuf   = (float*)carve((size_t)Bq * Lq * 4);
  float*     gsum   = (float*)carve((size_t)Bq * 4);
  _Float16*  wp16   = (_Float16*)carve((size_t)Dq * Pq * 2);
  _Float16*  h0w16  = (_Float16*)carve((size_t)PLq * KDq * 2);
  _Float16*  h1w116 = (_Float16*)carve((size_t)2 * PLq * KDq * 2);
  _Float16*  h1w216 = (_Float16*)carve((size_t)PLq * 2 * PLq * 2);
  _Float16*  pw16a  = (_Float16*)carve((size_t)PNq * PNq * 2);
  _Float16*  pw16b  = (_Float16*)carve((size_t)PNq * PNq * 2);
  _Float16*  zbuf   = (_Float16*)carve((size_t)NSq * KDq * 2);
  _Float16*  hA     = (_Float16*)carve((size_t)NSq * KDq * 2);
  _Float16*  hB     = (_Float16*)carve((size_t)NSq * KDq * 2);
  float*     ubuf   = (float*)carve((size_t)NSq * PLq * 4);
  _Float16*  v1h    = (_Float16*)carve((size_t)NSq * 2 * PLq * 2);
  _Float16*  pw16[2] = {pw16a, pw16b};

  // --- RevIN + normalize + gate projection ---
  stats_kernel<<<dim3(Bq, 3), 128, 0, stream>>>(x, meanB, sdevB, rstdB);
  norm_kernel<<<Bq * Lq, 128, 0, stream>>>(x, meanB, rstdB, revin_w, revin_b,
                                           start_w, start_b, xh, gbuf);
  gate_kernel<<<1, 32, 0, stream>>>(gbuf, gate_w, gate_b, gsum,
                                    (float*)d_out + (size_t)Bq * PLq * Cq);

  // --- weight f16 conversions ---
  cvt_kernel<<<(Dq * Pq + 255) / 256, 256, 0, stream>>>(wp_w, wp16, Dq * Pq);
  cvt_kernel<<<(PLq * KDq + 255) / 256, 256, 0, stream>>>(h0_w, h0w16, PLq * KDq);
  cvt_kernel<<<(2 * PLq * KDq + 255) / 256, 256, 0, stream>>>(h1_w1, h1w116, 2 * PLq * KDq);
  cvt_kernel<<<(PLq * 2 * PLq + 255) / 256, 256, 0, stream>>>(h1_w2, h1w216, PLq * 2 * PLq);
  cvt_kernel<<<(PNq * PNq + 255) / 256, 256, 0, stream>>>(pw_w[0], pw16[0], PNq * PNq);
  cvt_kernel<<<(PNq * PNq + 255) / 256, 256, 0, stream>>>(pw_w[1], pw16[1], PNq * PNq);

  // --- patch embedding z (WMMA) ---
  z_kernel<<<NSq, 128, 0, stream>>>(xh, wp16, wp_b, zbuf);

  // --- linear head u = z @ h0_w^T + h0_b (WMMA + TDM, N=96) ---
  gemm_nt_kernel<6, 0><<<41, 256, 0, stream>>>(zbuf, h0w16, h0_b, ubuf, nullptr,
                                               NSq, KDq);

  // --- mixer block 0 ---
  dwconv_kernel<<<NSq * PNq / 8, 256, 0, stream>>>(zbuf, hA, dw_w[0], dw_b[0],
                                                   bn1g[0], bn1b[0], bn1m[0], bn1v[0]);
  pw_kernel<<<NSq, 128, 0, stream>>>(hA, hB, pw16[0], pw_b[0],
                                     bn2g[0], bn2b[0], bn2m[0], bn2v[0]);
  // --- mixer block 1 (reuse zbuf as ping buffer) ---
  dwconv_kernel<<<NSq * PNq / 8, 256, 0, stream>>>(hB, zbuf, dw_w[1], dw_b[1],
                                                   bn1g[1], bn1b[1], bn1m[1], bn1v[1]);
  pw_kernel<<<NSq, 128, 0, stream>>>(zbuf, hA, pw16[1], pw_b[1],
                                     bn2g[1], bn2b[1], bn2m[1], bn2v[1]);

  // --- v = gelu(h @ h1_w1^T + b1) @ h1_w2^T + b2, accumulated into u ---
  gemm_nt_kernel<12, 1><<<41, 256, 0, stream>>>(hA, h1w116, h1_b1, nullptr, v1h,
                                                NSq, KDq);
  gemm_nt_kernel<6, 2><<<41, 256, 0, stream>>>(v1h, h1w216, h1_b2, ubuf, nullptr,
                                               NSq, 2 * PLq);

  // --- inverse RevIN + gate scale + transpose to (B, PL, C) ---
  final_kernel<<<(Bq * PLq * Cq + 255) / 256, 256, 0, stream>>>(
      ubuf, gsum, revin_w, revin_b, meanB, sdevB, (float*)d_out);
}
